// CFDiff_62457414419246
// MI455X (gfx1250) — matrix operand order repaired and checked by reading, not compiled, not run on hardware
//
#include <hip/hip_runtime.h>
#include <hip/hip_bf16.h>
#include <math.h>

typedef __attribute__((ext_vector_type(16))) __bf16 v16bf;
typedef __attribute__((ext_vector_type(8)))  float  v8f;

#define B_     1024
#define NI_    50000
#define NU_    20000
#define D_     256
#define H_     4
#define MAXNB_ 20
#define T_     1000
#define PI_F   3.14159265358979323846f

__device__ __forceinline__ float gelu_exact(float x) {
    return 0.5f * x * (1.0f + erff(x * 0.70710678f));
}
__device__ __forceinline__ float softplus_f(float x) {
    if (x > 20.f)  return x;
    if (x < -20.f) return __expf(x);
    return log1pf(__expf(x));
}

// ---------------------------------------------------------------------------
// WMMA bf16 GEMM:  C = act(A @ W^T + bias) [+ addv]   (A: MxK, W: NxK)
// BM=64, BN=64, BK=32; 256 threads = 8 waves; wave owns two 16x16 tiles.
// Double-buffered LDS pipeline: slice s+1 streamed by
// global_load_async_to_lds_b128 (ASYNCcnt) while WMMAs consume slice s;
// s_wait_asynccnt 4 drains only the previous slice's 4 per-wave transfers.
// Scalar guarded path for K-tail / partial-N tiles.
// RECON=true: accumulate softplus(c) - c*x0 into *acc instead of storing C.
// ---------------------------------------------------------------------------
template <int ACT, bool RECON>
__global__ __launch_bounds__(256) void gemm_wmma(
    const float* __restrict__ A, const float* __restrict__ W,
    const float* __restrict__ bias, const float* __restrict__ addv,
    float* __restrict__ C, int M, int N, int K,
    const float* __restrict__ x0, float* __restrict__ acc)
{
    constexpr int BM = 64, BN = 64, BK = 32;
    __shared__ float lds_A[2][BM * BK];
    __shared__ float lds_B[2][BN * BK];
    __shared__ float s_red;

    const int tid  = threadIdx.x;
    const int wave = tid >> 5;
    const int lane = tid & 31;
    const int lm   = lane & 15;        // M (A) or N (B,C) index within tile
    const int lk   = lane >> 4;        // lane-group selector

    const int mBase = blockIdx.y * BM;
    const int nBase = blockIdx.x * BN;
    const int wm = (wave & 3) * 16;    // wave's M sub-tile
    const int wn = (wave >> 2) * 32;   // wave's pair of N sub-tiles

    v8f c0 = {};
    v8f c1 = {};
    float lrecon = 0.f;
    if (RECON && tid == 0) s_red = 0.f;

    const bool fullN   = (nBase + BN) <= N;
    const int  nSlices = K / BK;        // # of full K slices
    const int  kTail   = nSlices * BK;

    auto issueA = [&](int k0, int buf) {
        #pragma unroll
        for (int r = 0; r < 2; ++r) {
            int e   = (tid + r * 256) * 4;
            int row = e >> 5, col = e & 31;
            unsigned long long ga =
                (unsigned long long)(const void*)(A + (size_t)(mBase + row) * K + k0 + col);
            unsigned la = (unsigned)(unsigned long long)(void*)&lds_A[buf][e];
            asm volatile("global_load_async_to_lds_b128 %0, %1, off"
                         :: "v"(la), "v"(ga) : "memory");
        }
    };
    auto issueB = [&](int k0, int buf) {
        #pragma unroll
        for (int r = 0; r < 2; ++r) {
            int e   = (tid + r * 256) * 4;
            int row = e >> 5, col = e & 31;
            unsigned long long gb =
                (unsigned long long)(const void*)(W + (size_t)(nBase + row) * K + k0 + col);
            unsigned lb = (unsigned)(unsigned long long)(void*)&lds_B[buf][e];
            asm volatile("global_load_async_to_lds_b128 %0, %1, off"
                         :: "v"(lb), "v"(gb) : "memory");
        }
    };
    auto compute = [&](int buf) {
        // A fragment: 16-bit A 16x32 layout (ISA 7.12.2): lanes 0-15 hold K
        // groups {0-7,16-23}, lanes 16-31 hold {8-15,24-31}, 2 vals per VGPR.
        const float* arow = &lds_A[buf][(wm + lm) * BK];
        v16bf a;
        #pragma unroll
        for (int j = 0; j < 8; ++j) {
            int kb = ((j >> 2) << 4) + (lk << 3) + ((j & 3) << 1);
            a[2 * j]     = (__bf16)arow[kb];
            a[2 * j + 1] = (__bf16)arow[kb + 1];
        }
        // B fragment: 32x16, same K order as A (lanes 0-15 {0-7,16-23}, 16-31 {8-15,24-31})
        const float* brow0 = &lds_B[buf][(wn + lm) * BK];
        const float* brow1 = &lds_B[buf][(wn + 16 + lm) * BK];
        v16bf b0, b1;
        #pragma unroll
        for (int j = 0; j < 8; ++j) {
            int kb = ((j >> 2) << 4) + (lk << 3) + ((j & 3) << 1);
            b0[2 * j]     = (__bf16)brow0[kb];
            b0[2 * j + 1] = (__bf16)brow0[kb + 1];
            b1[2 * j]     = (__bf16)brow1[kb];
            b1[2 * j + 1] = (__bf16)brow1[kb + 1];
        }
        c0 = __builtin_amdgcn_wmma_f32_16x16x32_bf16(false, a, false, b0,
                                                     (short)0, c0, false, false);
        c1 = __builtin_amdgcn_wmma_f32_16x16x32_bf16(false, a, false, b1,
                                                     (short)0, c1, false, false);
    };

    if (fullN && nSlices > 0) {
        // ---- double-buffered async pipeline over full K slices ----
        issueA(0, 0);
        issueB(0, 0);
        for (int s = 0; s < nSlices; ++s) {
            const int cur = s & 1;
            // prefetch streaming A operand ahead of the DMA window (L2 warm)
            if (s + 8 < nSlices)
                __builtin_prefetch(A + (size_t)(mBase + (tid & 63)) * K + (s + 8) * BK, 0, 1);
            if (s + 1 < nSlices) {
                issueA((s + 1) * BK, cur ^ 1);
                issueB((s + 1) * BK, cur ^ 1);
                asm volatile("s_wait_asynccnt 0x4" ::: "memory");  // prev slice done
            } else {
                asm volatile("s_wait_asynccnt 0x0" ::: "memory");
            }
            __syncthreads();
            compute(cur);
            __syncthreads();
        }
    } else {
        // ---- partial-N tile: scalar guarded loads, single buffer ----
        for (int s = 0; s < nSlices; ++s) {
            const int k0 = s * BK;
            for (int e = tid; e < BM * BK; e += 256) {
                int row = e >> 5, col = e & 31;
                lds_A[0][e] = A[(size_t)(mBase + row) * K + k0 + col];
            }
            for (int e = tid; e < BN * BK; e += 256) {
                int row = e >> 5, col = e & 31;
                int n = nBase + row;
                lds_B[0][e] = (n < N) ? W[(size_t)n * K + k0 + col] : 0.f;
            }
            __syncthreads();
            compute(0);
            __syncthreads();
        }
    }
    if (kTail < K) {
        // ---- K tail (< BK), zero-padded scalar loads ----
        for (int e = tid; e < BM * BK; e += 256) {
            int row = e >> 5, col = e & 31;
            int k = kTail + col;
            lds_A[0][e] = (k < K) ? A[(size_t)(mBase + row) * K + k] : 0.f;
        }
        for (int e = tid; e < BN * BK; e += 256) {
            int row = e >> 5, col = e & 31;
            int n = nBase + row, k = kTail + col;
            lds_B[0][e] = (n < N && k < K) ? W[(size_t)n * K + k] : 0.f;
        }
        __syncthreads();
        compute(0);
        __syncthreads();
    }

    // Epilogue. C layout: VGPR i -> M = i + 8*(lane>>4), N = lane&15.
    #pragma unroll
    for (int i = 0; i < 8; ++i) {
        int m  = mBase + wm + i + (lk << 3);
        int n0 = nBase + wn + lm;
        int n1 = n0 + 16;
        float v0 = c0[i] + bias[(n0 < N) ? n0 : 0];
        float v1 = c1[i] + bias[(n1 < N) ? n1 : 0];
        if (ACT == 1) { v0 = gelu_exact(v0); v1 = gelu_exact(v1); }
        if (RECON) {
            if (n0 < N) lrecon += softplus_f(v0) - v0 * x0[(size_t)m * N + n0];
            if (n1 < N) lrecon += softplus_f(v1) - v1 * x0[(size_t)m * N + n1];
        } else {
            if (addv) {
                if (n0 < N) v0 += addv[(size_t)m * N + n0];
                if (n1 < N) v1 += addv[(size_t)m * N + n1];
            }
            if (n0 < N) C[(size_t)m * N + n0] = v0;
            if (n1 < N) C[(size_t)m * N + n1] = v1;
        }
    }
    if (RECON) {
        atomicAdd(&s_red, lrecon);
        __syncthreads();
        if (tid == 0) atomicAdd(acc, s_red);
    }
}

// --- z_t = sqrt_ab[t]*z0 + sqrt_1ab[t]*noise ; te = silu(tn@w1+b1)@w2^T+b2 ---
__global__ __launch_bounds__(256) void zt_te_kernel(
    const float* __restrict__ z0, const float* __restrict__ noise,
    const int* __restrict__ t,
    const float* __restrict__ te_w1, const float* __restrict__ te_b1,
    const float* __restrict__ te_w2, const float* __restrict__ te_b2,
    float* __restrict__ zt, float* __restrict__ te)
{
    int b = blockIdx.x, tid = threadIdx.x;
    __shared__ float h32[32];
    __shared__ float sa_s, sb_s;
    int tt = t[b];
    if (tid == 0) {
        float s  = (float)(tt + 1) / (float)T_;
        float c  = cosf((s + 0.008f) / 1.008f * PI_F * 0.5f);
        float c0 = cosf((0.008f / 1.008f) * PI_F * 0.5f);
        float r  = (c * c) / (c0 * c0);
        r = fminf(fmaxf(r, 0.f), 1.f);
        sa_s = sqrtf(r);
        sb_s = sqrtf(1.f - r);
    }
    if (tid < 32) {
        float tn = (float)tt / (float)T_;
        float hv = tn * te_w1[tid] + te_b1[tid];
        h32[tid] = hv / (1.f + __expf(-hv));   // silu
    }
    __syncthreads();
    float accv = te_b2[tid];
    #pragma unroll 8
    for (int j = 0; j < 32; ++j) accv += te_w2[tid * 32 + j] * h32[j];
    size_t idx = (size_t)b * D_ + tid;
    te[idx] = accv;
    zt[idx] = sa_s * z0[idx] + sb_s * noise[idx];
}

// --- nb gather: nb[b,n,:] = item_emb[neighbor_idx[user_ids[b], n], :] ---
__global__ void gather_kernel(const int* __restrict__ user_ids,
                              const int* __restrict__ neighbor_idx,
                              const float* __restrict__ item_emb,
                              float* __restrict__ nb)
{
    size_t i = (size_t)blockIdx.x * blockDim.x + threadIdx.x;
    size_t total = (size_t)B_ * MAXNB_ * D_;
    if (i >= total) return;
    int d = (int)(i & (D_ - 1));
    size_t row = i >> 8;
    int n = (int)(row % MAXNB_);
    int b = (int)(row / MAXNB_);
    int item = neighbor_idx[(size_t)user_ids[b] * MAXNB_ + n];
    nb[i] = item_emb[(size_t)item * D_ + d];
}

// --- cross-attention core: 1 query, 20 keys, 4 heads of 64 ---
__global__ __launch_bounds__(256) void attn_kernel(
    const float* __restrict__ qp, const float* __restrict__ kp,
    const float* __restrict__ vp, float* __restrict__ out)
{
    int b = blockIdx.x, tid = threadIdx.x;
    __shared__ float qs[D_];
    __shared__ float att[H_][MAXNB_];
    qs[tid] = qp[(size_t)b * D_ + tid];
    __syncthreads();
    if (tid < H_ * MAXNB_) {
        int h = tid / MAXNB_, k = tid % MAXNB_;
        const float* kr = kp + ((size_t)b * MAXNB_ + k) * D_ + h * 64;
        const float* qr = qs + h * 64;
        float s = 0.f;
        #pragma unroll 8
        for (int d = 0; d < 64; ++d) s += qr[d] * kr[d];
        att[h][k] = s * 0.125f;    // 1/sqrt(64)
    }
    __syncthreads();
    if (tid < H_) {
        float mx = -1e30f;
        for (int k = 0; k < MAXNB_; ++k) mx = fmaxf(mx, att[tid][k]);
        float ssum = 0.f;
        for (int k = 0; k < MAXNB_; ++k) {
            float e = __expf(att[tid][k] - mx);
            att[tid][k] = e; ssum += e;
        }
        float inv = 1.f / ssum;
        for (int k = 0; k < MAXNB_; ++k) att[tid][k] *= inv;
    }
    __syncthreads();
    int h = tid >> 6;
    float o = 0.f;
    #pragma unroll
    for (int k = 0; k < MAXNB_; ++k)
        o += att[h][k] * vp[((size_t)b * MAXNB_ + k) * D_ + tid];
    out[(size_t)b * D_ + tid] = o;
}

// --- LayerNorm over D=256, one block per row (wave32 shuffles) ---
__global__ __launch_bounds__(256) void ln_kernel(
    const float* __restrict__ x, const float* __restrict__ g,
    const float* __restrict__ bb, float* __restrict__ out)
{
    int b = blockIdx.x, tid = threadIdx.x;
    __shared__ float rs[8], rq[8];
    __shared__ float mean_s, inv_s;
    float v = x[(size_t)b * D_ + tid];
    float s = v, q = v * v;
    #pragma unroll
    for (int o = 16; o > 0; o >>= 1) {
        s += __shfl_down(s, o, 32);
        q += __shfl_down(q, o, 32);
    }
    if ((tid & 31) == 0) { rs[tid >> 5] = s; rq[tid >> 5] = q; }
    __syncthreads();
    if (tid == 0) {
        float S = 0.f, Q = 0.f;
        for (int i = 0; i < 8; ++i) { S += rs[i]; Q += rq[i]; }
        float m = S / (float)D_;
        mean_s = m;
        inv_s  = rsqrtf(Q / (float)D_ - m * m + 1e-5f);
    }
    __syncthreads();
    out[(size_t)b * D_ + tid] = (v - mean_s) * inv_s * g[tid] + bb[tid];
}

__global__ void diff_loss_kernel(const float* __restrict__ zp,
                                 const float* __restrict__ z0,
                                 float* __restrict__ acc)
{
    __shared__ float s_red;
    if (threadIdx.x == 0) s_red = 0.f;
    __syncthreads();
    size_t total = (size_t)B_ * D_;
    float l = 0.f;
    for (size_t i = (size_t)blockIdx.x * 256 + threadIdx.x; i < total;
         i += (size_t)gridDim.x * 256) {
        float d = zp[i] - z0[i];
        l += d * d;
    }
    atomicAdd(&s_red, l);
    __syncthreads();
    if (threadIdx.x == 0) atomicAdd(acc, s_red);
}

__global__ void zero_kernel(float* acc) {
    if (threadIdx.x < 2) acc[threadIdx.x] = 0.f;
}

__global__ void final_kernel(const float* __restrict__ acc, float* __restrict__ out) {
    if (threadIdx.x == 0)
        out[0] = acc[0] / ((float)B_ * (float)D_) +
                 0.1f * acc[1] / ((float)B_ * (float)NI_);
}

// ---------------------------------------------------------------------------
extern "C" void kernel_launch(void* const* d_in, const int* in_sizes, int n_in,
                              void* d_out, int out_size, void* d_ws, size_t ws_size,
                              hipStream_t stream)
{
    const float* x0       = (const float*)d_in[0];
    const int*   user_ids = (const int*)  d_in[1];
    const int*   t        = (const int*)  d_in[2];
    const float* noise    = (const float*)d_in[3];
    const int*   nbr_idx  = (const int*)  d_in[4];
    const float* item_emb = (const float*)d_in[5];
    const float* enc_w1   = (const float*)d_in[6];
    const float* enc_b1   = (const float*)d_in[7];
    const float* enc_w2   = (const float*)d_in[8];
    const float* enc_b2   = (const float*)d_in[9];
    const float* dec_w1   = (const float*)d_in[10];
    const float* dec_b1   = (const float*)d_in[11];
    const float* dec_w2   = (const float*)d_in[12];
    const float* dec_b2   = (const float*)d_in[13];
    const float* up_w     = (const float*)d_in[14];
    const float* up_b     = (const float*)d_in[15];
    const float* ip_w     = (const float*)d_in[16];
    const float* ip_b     = (const float*)d_in[17];
    const float* te_w1    = (const float*)d_in[18];
    const float* te_b1    = (const float*)d_in[19];
    const float* te_w2    = (const float*)d_in[20];
    const float* te_b2    = (const float*)d_in[21];
    const float* ca_wqkv  = (const float*)d_in[22];
    const float* ca_bqkv  = (const float*)d_in[23];
    const float* ca_wo    = (const float*)d_in[24];
    const float* ca_bo    = (const float*)d_in[25];
    const float* sa_wqkv  = (const float*)d_in[26];
    const float* sa_bqkv  = (const float*)d_in[27];
    const float* sa_wo    = (const float*)d_in[28];
    const float* sa_bo    = (const float*)d_in[29];
    const float* n1_g     = (const float*)d_in[30];
    const float* n1_b     = (const float*)d_in[31];
    const float* n2_g     = (const float*)d_in[32];
    const float* n2_b     = (const float*)d_in[33];
    const float* ff_w1    = (const float*)d_in[34];
    const float* ff_b1    = (const float*)d_in[35];
    const float* ff_w2    = (const float*)d_in[36];
    const float* ff_b2    = (const float*)d_in[37];

    float* w = (float*)d_ws;
    size_t o = 0;
    auto alloc = [&](size_t n) { float* p = w + o; o += n; return p; };
    float* acc   = alloc(2);
    float* h1    = alloc((size_t)B_ * 2 * D_);           // 1024x512
    float* z0    = alloc((size_t)B_ * D_);
    float* zt    = alloc((size_t)B_ * D_);
    float* te    = alloc((size_t)B_ * D_);
    float* q     = alloc((size_t)B_ * D_);
    float* nb    = alloc((size_t)B_ * MAXNB_ * D_);      // reused as kp
    float* kv    = alloc((size_t)B_ * MAXNB_ * D_);
    float* vp    = alloc((size_t)B_ * MAXNB_ * D_);
    float* qp    = alloc((size_t)B_ * D_);
    float* ao    = alloc((size_t)B_ * D_);
    float* capre = alloc((size_t)B_ * D_);
    float* h     = alloc((size_t)B_ * D_);
    float* vprj  = alloc((size_t)B_ * D_);
    float* h2pre = alloc((size_t)B_ * D_);
    float* h2    = alloc((size_t)B_ * D_);
    float* ff1   = alloc((size_t)B_ * 2 * D_);
    float* zp    = alloc((size_t)B_ * D_);
    float* dech  = alloc((size_t)B_ * 2 * D_);
    (void)ws_size; (void)n_in; (void)in_sizes; (void)out_size;

    auto grid = [](int M, int N) { return dim3((N + 63) / 64, M / 64); };

    zero_kernel<<<1, 32, 0, stream>>>(acc);

    // z0 = gelu(x0 @ enc_w1^T + b1) @ enc_w2^T + b2
    gemm_wmma<1, false><<<grid(B_, 2 * D_), 256, 0, stream>>>(
        x0, enc_w1, enc_b1, nullptr, h1, B_, 2 * D_, NI_, nullptr, nullptr);
    gemm_wmma<0, false><<<grid(B_, D_), 256, 0, stream>>>(
        h1, enc_w2, enc_b2, nullptr, z0, B_, D_, 2 * D_, nullptr, nullptr);

    // z_t, time embedding
    zt_te_kernel<<<B_, 256, 0, stream>>>(z0, noise, t, te_w1, te_b1, te_w2, te_b2, zt, te);

    // neighbor gather + input projection
    size_t nbelem = (size_t)B_ * MAXNB_ * D_;
    gather_kernel<<<(int)((nbelem + 255) / 256), 256, 0, stream>>>(user_ids, nbr_idx, item_emb, nb);
    gemm_wmma<0, false><<<grid(B_ * MAXNB_, D_), 256, 0, stream>>>(
        nb, ip_w, ip_b, nullptr, kv, B_ * MAXNB_, D_, D_, nullptr, nullptr);

    // q = z_t @ up_w^T + up_b + te
    gemm_wmma<0, false><<<grid(B_, D_), 256, 0, stream>>>(
        zt, up_w, up_b, te, q, B_, D_, D_, nullptr, nullptr);

    // cross-attn projections (wq/wk/wv are row-blocks of ca_wqkv)
    gemm_wmma<0, false><<<grid(B_, D_), 256, 0, stream>>>(
        q, ca_wqkv, ca_bqkv, nullptr, qp, B_, D_, D_, nullptr, nullptr);
    gemm_wmma<0, false><<<grid(B_ * MAXNB_, D_), 256, 0, stream>>>(
        kv, ca_wqkv + (size_t)D_ * D_, ca_bqkv + D_, nullptr, nb,  // kp -> nb
        B_ * MAXNB_, D_, D_, nullptr, nullptr);
    gemm_wmma<0, false><<<grid(B_ * MAXNB_, D_), 256, 0, stream>>>(
        kv, ca_wqkv + (size_t)2 * D_ * D_, ca_bqkv + 2 * D_, nullptr, vp,
        B_ * MAXNB_, D_, D_, nullptr, nullptr);

    attn_kernel<<<B_, 256, 0, stream>>>(qp, nb, vp, ao);

    // ca out-proj + residual, LN1
    gemm_wmma<0, false><<<grid(B_, D_), 256, 0, stream>>>(
        ao, ca_wo, ca_bo, q, capre, B_, D_, D_, nullptr, nullptr);
    ln_kernel<<<B_, 256, 0, stream>>>(capre, n1_g, n1_b, h);

    // self-attn with 1 token: softmax==1 -> sa = (h@wv^T+bv)@wo^T+bo (exact)
    gemm_wmma<0, false><<<grid(B_, D_), 256, 0, stream>>>(
        h, sa_wqkv + (size_t)2 * D_ * D_, sa_bqkv + 2 * D_, nullptr, vprj,
        B_, D_, D_, nullptr, nullptr);
    gemm_wmma<0, false><<<grid(B_, D_), 256, 0, stream>>>(
        vprj, sa_wo, sa_bo, h, h2pre, B_, D_, D_, nullptr, nullptr);
    ln_kernel<<<B_, 256, 0, stream>>>(h2pre, n2_g, n2_b, h2);

    // FFN + residual
    gemm_wmma<1, false><<<grid(B_, 2 * D_), 256, 0, stream>>>(
        h2, ff_w1, ff_b1, nullptr, ff1, B_, 2 * D_, D_, nullptr, nullptr);
    gemm_wmma<0, false><<<grid(B_, D_), 256, 0, stream>>>(
        ff1, ff_w2, ff_b2, h2, zp, B_, D_, 2 * D_, nullptr, nullptr);

    diff_loss_kernel<<<256, 256, 0, stream>>>(zp, z0, acc);

    // decoder: dech = gelu(z0@dec_w1^T+b1); fused logits+recon reduction
    gemm_wmma<1, false><<<grid(B_, 2 * D_), 256, 0, stream>>>(
        z0, dec_w1, dec_b1, nullptr, dech, B_, 2 * D_, D_, nullptr, nullptr);
    gemm_wmma<0, true><<<grid(B_, NI_), 256, 0, stream>>>(
        dech, dec_w2, dec_b2, nullptr, nullptr, B_, NI_, 2 * D_, x0, acc + 1);

    final_kernel<<<1, 32, 0, stream>>>(acc, (float*)d_out);
}
